// DecoderTT_15229954032088
// MI455X (gfx1250) — matrix operand order repaired and checked by reading, not compiled, not run on hardware
//
#include <hip/hip_runtime.h>

typedef __attribute__((ext_vector_type(16))) _Float16 v16h;
typedef __attribute__((ext_vector_type(4)))  _Float16 v4h;
typedef __attribute__((ext_vector_type(8)))  float    v8f;

#define BDIM 2
#define TDIM 512
#define UDIM 128
#define EDIM 512   // == DDIM
#define JDIM 640
#define ODIM 600

#define KSTEPS (JDIM / 32)   // 20 k-steps of 32
#define NTILES 38            // ceil(600/16), padded to 608
#define MTILES (UDIM / 16)   // 8

// LDS: A-fragments for whole z panel: [MTILES][KSTEPS][32 lanes][16 halves]
#define ZSMEM_BYTES (MTILES * KSTEPS * 32 * 16 * 2)   // 163840 B (<= 320KB WGP LDS)

// CDNA5 transcendental tanh (trans op); v_nop covers the 1-op trans hazard.
__device__ __forceinline__ float fast_tanh(float x) {
  float r;
  asm volatile("v_tanh_f32 %0, %1\n\tv_nop" : "=v"(r) : "v"(x));
  return r;
}

// ---------------------------------------------------------------------------
// Kernel 1: pe[b,t,j] = h_enc . W_enc[j] + b_enc[j] ; pd[b,u,j] = h_dec . W_dec[j]
// rows 0..1023 -> pe, rows 1024..1279 -> pd. One block per row.
// ---------------------------------------------------------------------------
__global__ __launch_bounds__(256) void proj_kernel(
    const float* __restrict__ h_enc, const float* __restrict__ h_dec,
    const float* __restrict__ W_enc, const float* __restrict__ b_enc,
    const float* __restrict__ W_dec,
    float* __restrict__ pe, float* __restrict__ pd) {
  __shared__ float xs[EDIM];
  const int r   = blockIdx.x;            // 0..1279
  const int tid = threadIdx.x;
  const bool isE = r < (BDIM * TDIM);

  const float* x;
  const float* W;
  float* dst;
  if (isE) {
    x = h_enc + (size_t)r * EDIM;
    W = W_enc;
    dst = pe + (size_t)r * JDIM;
  } else {
    const int rr = r - BDIM * TDIM;
    x = h_dec + (size_t)rr * EDIM;
    W = W_dec;
    dst = pd + (size_t)rr * JDIM;
  }

  if (tid < EDIM / 4)
    reinterpret_cast<float4*>(xs)[tid] = reinterpret_cast<const float4*>(x)[tid];
  __syncthreads();

  for (int j = tid; j < JDIM; j += 256) {
    const float4* w4 = reinterpret_cast<const float4*>(W + (size_t)j * EDIM);
    const float4* x4 = reinterpret_cast<const float4*>(xs);
    float acc = isE ? b_enc[j] : 0.0f;
#pragma unroll 8
    for (int k = 0; k < EDIM / 4; ++k) {
      const float4 wv = w4[k];
      const float4 xv = x4[k];
      acc += wv.x * xv.x + wv.y * xv.y + wv.z * xv.z + wv.w * xv.w;
    }
    dst[j] = acc;
  }
}

// ---------------------------------------------------------------------------
// Kernel 2: pack W_out (600x640 f32) into f16 WMMA B-fragments, zero-padded to
// 608 columns. Fragment (nt, ks): lane L holds n = nt*16 + (L&15),
// k = ks*32 + (L<16?0:16) + e  for e = 0..15.
// ---------------------------------------------------------------------------
__global__ __launch_bounds__(32) void pack_wout_kernel(
    const float* __restrict__ W_out, _Float16* __restrict__ Bp) {
  const int idx  = blockIdx.x;           // nt * KSTEPS + ks, 0..759
  const int nt   = idx / KSTEPS;
  const int ks   = idx - nt * KSTEPS;
  const int lane = threadIdx.x;
  const int o    = nt * 16 + (lane & 15);
  const int jb   = ks * 32 + ((lane & 16) ? 16 : 0);
  v16h f;
#pragma unroll
  for (int e = 0; e < 16; ++e) {
    const float v = (o < ODIM) ? W_out[(size_t)o * JDIM + jb + e] : 0.0f;
    f[e] = (_Float16)v;
  }
  reinterpret_cast<v16h*>(Bp)[idx * 32 + lane] = f;
}

// ---------------------------------------------------------------------------
// Kernel 3: fused tanh(pe + pd) -> GEMM with W_out. One WG per (b,t).
// Phase 1: build z (128 x 640) in f16 A-fragment layout in LDS (tanh once).
// Phase 2: 8 waves = 4 M-pairs x 2 N-halves. Each wave owns 2 M-tiles and
//          every other N-tile: B fragment loaded once per k, fed to 2 WMMAs.
// ---------------------------------------------------------------------------
__global__ __launch_bounds__(256) void joint_kernel(
    const float* __restrict__ pe, const float* __restrict__ pd,
    const _Float16* __restrict__ Bp, const float* __restrict__ b_out,
    float* __restrict__ out) {
  extern __shared__ float4 smem4[];
  _Float16* z = reinterpret_cast<_Float16*>(smem4);  // [MTILES][KSTEPS][32][16]

  const int bt  = blockIdx.x;            // b*T + t
  const int b   = bt / TDIM;
  const int tid = threadIdx.x;

  const float* peRow = pe + (size_t)bt * JDIM;
  const float* pdMat = pd + (size_t)b * UDIM * JDIM;

  // ---- Phase 1: z[u][j] = tanh(pe[j] + pd[u][j]) -> f16 A-fragments in LDS
  for (int it = tid; it < UDIM * (JDIM / 4); it += 256) {
    const int u  = it / (JDIM / 4);
    const int jv = it - u * (JDIM / 4);
    const int j0 = jv << 2;

    const float4 a = *reinterpret_cast<const float4*>(peRow + j0);
    const float4 d = *reinterpret_cast<const float4*>(pdMat + (size_t)u * JDIM + j0);
    v4h h;
    h[0] = (_Float16)fast_tanh(a.x + d.x);
    h[1] = (_Float16)fast_tanh(a.y + d.y);
    h[2] = (_Float16)fast_tanh(a.z + d.z);
    h[3] = (_Float16)fast_tanh(a.w + d.w);

    // A-fragment slot: lane = (u&15) | (j0&16); e0 = j0&15
    const int mT   = u >> 4;
    const int kS   = j0 >> 5;
    const int lane = (u & 15) | (j0 & 16);
    const int e0   = j0 & 15;     // {0,4,8,12}: 8B aligned
    const int hIdx = (((mT * KSTEPS + kS) << 5) + lane) * 16 + e0;
    *reinterpret_cast<v4h*>(z + hIdx) = h;
  }
  __syncthreads();

  // ---- Phase 2: wave (mh, nh): mTiles {2mh, 2mh+1}, nt = nh, nh+2, ...
  const int w    = tid >> 5;
  const int lane = tid & 31;
  const int mh   = w >> 1;               // 0..3
  const int nh   = w & 1;                // 0..1
  const int mT0  = mh << 1;              // first of the two M-tiles

  const _Float16* ap0 = z + (size_t)((mT0 * KSTEPS) * 32 + lane) * 16;
  const _Float16* ap1 = ap0 + (size_t)KSTEPS * 32 * 16;      // mT0+1
  const v16h* bbase   = reinterpret_cast<const v16h*>(Bp);
  const int ncol = lane & 15;
  const int m0   = (lane & 16) ? 8 : 0;

  // output row base for mTile mT0 (c0); c1 rows are +16 u's
  float* outR0 = out + ((size_t)bt * UDIM + mT0 * 16 + m0) * ODIM;
  float* outR1 = outR0 + (size_t)16 * ODIM;

  for (int nt = nh; nt < NTILES; nt += 2) {
    const int o = nt * 16 + ncol;                 // up to 607 on last tile
    const bool valid = (o < ODIM);
    const float bias = valid ? b_out[o] : 0.0f;
    v8f c0, c1;
#pragma unroll
    for (int v = 0; v < 8; ++v) { c0[v] = bias; c1[v] = bias; }

    const v16h* bp = bbase + (nt * KSTEPS) * 32 + lane;
#pragma unroll 4
    for (int k = 0; k < KSTEPS; ++k) {
      const v16h Bf = bp[k * 32];
      const v16h A0 = *reinterpret_cast<const v16h*>(ap0 + (size_t)k * 512);
      const v16h A1 = *reinterpret_cast<const v16h*>(ap1 + (size_t)k * 512);
      c0 = __builtin_amdgcn_wmma_f32_16x16x32_f16(false, A0, false, Bf,
                                                  (short)0, c0, false, false);
      c1 = __builtin_amdgcn_wmma_f32_16x16x32_f16(false, A1, false, Bf,
                                                  (short)0, c1, false, false);
    }

    if (valid) {
#pragma unroll
      for (int v = 0; v < 8; ++v) {
        outR0[(size_t)v * ODIM + o] = c0[v];
        outR1[(size_t)v * ODIM + o] = c1[v];
      }
    }
  }
}

// ---------------------------------------------------------------------------
extern "C" void kernel_launch(void* const* d_in, const int* in_sizes, int n_in,
                              void* d_out, int out_size, void* d_ws, size_t ws_size,
                              hipStream_t stream) {
  const float* h_enc = (const float*)d_in[0];
  const float* h_dec = (const float*)d_in[1];
  const float* W_enc = (const float*)d_in[2];
  const float* b_enc = (const float*)d_in[3];
  const float* W_dec = (const float*)d_in[4];
  const float* W_out = (const float*)d_in[5];
  const float* b_out = (const float*)d_in[6];
  float* out = (float*)d_out;

  // Workspace layout (needs ~4.06 MB):
  float* pe = (float*)d_ws;                                    // 1024*640 f32
  float* pd = pe + (size_t)BDIM * TDIM * JDIM;                 //  256*640 f32
  _Float16* Bp = (_Float16*)(pd + (size_t)BDIM * UDIM * JDIM); // 760*32*16 f16

  proj_kernel<<<BDIM * TDIM + BDIM * UDIM, 256, 0, stream>>>(
      h_enc, h_dec, W_enc, b_enc, W_dec, pe, pd);

  pack_wout_kernel<<<NTILES * KSTEPS, 32, 0, stream>>>(W_out, Bp);

  hipFuncSetAttribute(reinterpret_cast<const void*>(joint_kernel),
                      hipFuncAttributeMaxDynamicSharedMemorySize, ZSMEM_BYTES);
  joint_kernel<<<BDIM * TDIM, 256, ZSMEM_BYTES, stream>>>(pe, pd, Bp, b_out, out);
}
